// MultiViewAttention_69131793596834
// MI455X (gfx1250) — matrix operand order repaired and checked by reading, not compiled, not run on hardware
//
#include <hip/hip_runtime.h>

typedef __attribute__((ext_vector_type(16))) _Float16 v16h;
typedef __attribute__((ext_vector_type(8)))  float    v8f;
typedef __attribute__((ext_vector_type(4)))  float    f32x4;

#define C_DIM 256
#define HW    1024
#define NH    8
#define HD    32

union AF  { v16h v; _Float16 h[16]; f32x4 f4[2]; };
union H8  { _Float16 h[8];  f32x4 f4; };

// ---------------------------------------------------------------------------
// Prep: f32 -> f16 weight copies (Wq,Wk,Wv,Wp packed at slots 0..3)
// ---------------------------------------------------------------------------
__global__ void prep_weights(const float* __restrict__ Wq, const float* __restrict__ Wk,
                             const float* __restrict__ Wv, const float* __restrict__ Wp,
                             _Float16* __restrict__ dst) {
  int i = blockIdx.x * 256 + threadIdx.x;          // 0 .. 4*65536-1
  int m = i >> 16, off = i & 0xFFFF;
  const float* src = (m == 0) ? Wq : (m == 1) ? Wk : (m == 2) ? Wv : Wp;
  dst[i] = (_Float16)src[off];
}

// x [B][C][HW] f32  ->  xt [B][HW][C] f16 (pixel-major for B-fragments)
__global__ void prep_x(const float* __restrict__ x, _Float16* __restrict__ xt) {
  int i = blockIdx.x * 256 + threadIdx.x;          // 0 .. 2M-1, i = b*HW*C + p*C + c
  int c = i & 255, p = (i >> 8) & 1023, b = i >> 18;
  xt[i] = (_Float16)x[((size_t)b * C_DIM + c) * HW + p];
}

// ---------------------------------------------------------------------------
// QKV: per wave one 16(ch) x 64(px) strip of W[256x256] @ x[256x1024] (+bias).
// The 8 K-step A-fragments (W rows) are register-resident across the 4 pixel
// sub-tiles: 80 b128 loads per 32 WMMAs.
// Q,K -> pixel-major [B][HW][C] f16 ; V -> d-major [B][C][HW] f16
// ---------------------------------------------------------------------------
__global__ void qkv_kernel(const _Float16* __restrict__ Wf16, const _Float16* __restrict__ xt,
                           const float* __restrict__ bq, const float* __restrict__ bk,
                           const float* __restrict__ bv,
                           _Float16* __restrict__ qt, _Float16* __restrict__ kt,
                           _Float16* __restrict__ vdm) {
  int wave = threadIdx.x >> 5, lane = threadIdx.x & 31;
  int hl = lane >> 4, ln = lane & 15;
  int t = blockIdx.x * 4 + wave;                   // 0 .. 6143
  int pg   = t & 15;                               // 64-pixel group
  int ot   = (t >> 4) & 15;                        // 16-channel tile
  int proj = (t >> 8) % 3;                         // 0=q 1=k 2=v
  int b    = t / 768;
  int pbase = pg * 64;

  const _Float16* W  = Wf16 + (size_t)proj * 65536;
  const _Float16* xb = xt + (size_t)b * HW * C_DIM;
  const float* bias = (proj == 0) ? bq : (proj == 1) ? bk : bv;

  // Hoist the full K=256 strip of W A-fragments into registers (8 x v16h).
  AF aw[8];
  for (int ks = 0; ks < 8; ++ks) {
    const _Float16* wr = W + (size_t)(ot * 16 + ln) * C_DIM + ks * 32 + hl * 8;
    aw[ks].f4[0] = *(const f32x4*)(wr);
    aw[ks].f4[1] = *(const f32x4*)(wr + 16);
  }

  float brow[8];
  for (int r = 0; r < 8; ++r) brow[r] = bias[ot * 16 + r + 8 * hl];

  for (int j = 0; j < 4; ++j) {                    // 4 pixel sub-tiles
    int ptile = pbase + j * 16;
    v8f acc = {};
    for (int ks = 0; ks < 8; ++ks) {
      AF bf;
      const _Float16* xr = xb + (size_t)(ptile + ln) * C_DIM + ks * 32 + hl * 8;
      bf.f4[0] = *(const f32x4*)(xr);
      bf.f4[1] = *(const f32x4*)(xr + 16);
      acc = __builtin_amdgcn_wmma_f32_16x16x32_f16(false, aw[ks].v, false, bf.v,
                                                   (short)0, acc, false, false);
    }
    if (proj < 2) {                                // q/k: pixel-major, 16B packed store
      _Float16* dst = (proj == 0) ? qt : kt;
      H8 pk;
      for (int r = 0; r < 8; ++r) pk.h[r] = (_Float16)(acc[r] + brow[r]);
      *(f32x4*)(dst + ((size_t)b * HW + ptile + ln) * C_DIM + ot * 16 + 8 * hl) = pk.f4;
    } else {                                       // v: d-major (strided f16 stores)
      for (int r = 0; r < 8; ++r)
        vdm[((size_t)b * C_DIM + ot * 16 + r + 8 * hl) * HW + ptile + ln] =
            (_Float16)(acc[r] + brow[r]);
    }
  }
}

// ---------------------------------------------------------------------------
// Attention: per wave one (b, h, 16-query tile). Scores strip in LDS (f16),
// softmax via half-wave shfl_xor(16), P@V^T with WMMA, 1/sum via bpermute.
// ---------------------------------------------------------------------------
__global__ void attn_kernel(const _Float16* __restrict__ qt, const _Float16* __restrict__ kt,
                            const _Float16* __restrict__ vdm, _Float16* __restrict__ attnP) {
  extern __shared__ _Float16 smem[];               // 4 waves * 16*1032 halves
  int wave = threadIdx.x >> 5, lane = threadIdx.x & 31;
  int hl = lane >> 4, ln = lane & 15;
  int t = blockIdx.x * 4 + wave;                   // 0 .. 4095
  int qtile = t & 63;
  int h = (t >> 6) & 7;
  int b = t >> 9;
  _Float16* S = smem + (size_t)wave * 16 * 1032;   // row stride 1032 breaks bank align

  // Q A-fragment (16 queries x 32 dims), loaded once
  AF aq;
  const _Float16* qrow = qt + ((size_t)b * HW + qtile * 16 + ln) * C_DIM + h * HD + hl * 8;
  aq.f4[0] = *(const f32x4*)(qrow);
  aq.f4[1] = *(const f32x4*)(qrow + 16);

  const float scale = 0.17677669529663687f;        // 1/sqrt(32)

  // Phase 1: S = Q^T K, scaled, stored f16 in LDS
  for (int jt = 0; jt < 64; ++jt) {
    const _Float16* krow = kt + ((size_t)b * HW + jt * 16 + ln) * C_DIM + h * HD + hl * 8;
    if (jt < 63) __builtin_prefetch(krow + C_DIM * 16, 0, 3);
    AF bk;
    bk.f4[0] = *(const f32x4*)(krow);
    bk.f4[1] = *(const f32x4*)(krow + 16);
    v8f c = {};
    c = __builtin_amdgcn_wmma_f32_16x16x32_f16(false, aq.v, false, bk.v,
                                               (short)0, c, false, false);
    for (int r = 0; r < 8; ++r)
      S[(size_t)(r + 8 * hl) * 1032 + jt * 16 + ln] = (_Float16)(c[r] * scale);
  }
  __syncthreads();

  // Phase 2: row softmax. Lane handles row ln, columns hl*512 .. +511.
  _Float16* row = S + (size_t)ln * 1032 + hl * 512;
  float m = -1e30f;
  for (int i = 0; i < 64; ++i) {
    H8 u; u.f4 = *(const f32x4*)(row + i * 8);
    for (int k = 0; k < 8; ++k) m = fmaxf(m, (float)u.h[k]);
  }
  m = fmaxf(m, __shfl_xor(m, 16, 32));
  float s = 0.f;
  for (int i = 0; i < 64; ++i) {
    H8 u; u.f4 = *(const f32x4*)(row + i * 8);
    for (int k = 0; k < 8; ++k) {
      float e = __expf((float)u.h[k] - m);
      s += e;
      u.h[k] = (_Float16)e;                        // un-normalized prob, in place
    }
    *(f32x4*)(row + i * 8) = u.f4;
  }
  s += __shfl_xor(s, 16, 32);
  float inv = 1.0f / s;
  __syncthreads();

  // Phase 3: O = P @ V^T (two 16-wide N tiles covering hd=32)
  v8f o0 = {}, o1 = {};
  const _Float16* v0base = vdm + ((size_t)b * C_DIM + h * HD + ln) * HW;
  const _Float16* v1base = v0base + (size_t)16 * HW;
  for (int jc = 0; jc < 32; ++jc) {
    AF ap, bv0, bv1;
    const _Float16* pr = S + (size_t)ln * 1032 + jc * 32 + hl * 8;
    ap.f4[0] = *(const f32x4*)(pr);
    ap.f4[1] = *(const f32x4*)(pr + 16);
    const _Float16* vp0 = v0base + jc * 32 + hl * 8;
    bv0.f4[0] = *(const f32x4*)(vp0);
    bv0.f4[1] = *(const f32x4*)(vp0 + 16);
    const _Float16* vp1 = v1base + jc * 32 + hl * 8;
    bv1.f4[0] = *(const f32x4*)(vp1);
    bv1.f4[1] = *(const f32x4*)(vp1 + 16);
    o0 = __builtin_amdgcn_wmma_f32_16x16x32_f16(false, ap.v, false, bv0.v,
                                                (short)0, o0, false, false);
    o1 = __builtin_amdgcn_wmma_f32_16x16x32_f16(false, ap.v, false, bv1.v,
                                                (short)0, o1, false, false);
  }

  // Phase 4: normalize rows (inv for row r+8*hl lives in lane r+8*hl) and store
  _Float16* obase = attnP + ((size_t)b * HW + qtile * 16) * C_DIM + h * HD;
  for (int r = 0; r < 8; ++r) {
    float invr = __shfl(inv, r + 8 * hl, 32);
    int pix = r + 8 * hl;
    obase[(size_t)pix * C_DIM + ln]      = (_Float16)(o0[r] * invr);
    obase[(size_t)pix * C_DIM + 16 + ln] = (_Float16)(o1[r] * invr);
  }
}

// ---------------------------------------------------------------------------
// Projection + residual + channel LayerNorm. One WG = 16 pixels x 256 ch.
// The 8 attnP B-fragments are register-resident across a wave's 4 ch tiles.
// ---------------------------------------------------------------------------
__global__ void proj_ln_kernel(const _Float16* __restrict__ Wpf16,
                               const _Float16* __restrict__ attnP,
                               const float* __restrict__ bp, const float* __restrict__ x,
                               const float* __restrict__ gamma, const float* __restrict__ beta,
                               float* __restrict__ out) {
  __shared__ float ybuf[16 * 264];
  __shared__ float psum[128], psq[128];
  __shared__ float mean_s[16], inv_s[16];
  int wave = threadIdx.x >> 5, lane = threadIdx.x & 31;
  int hl = lane >> 4, ln = lane & 15;
  int b = blockIdx.x >> 6;
  int pbase = (blockIdx.x & 63) * 16;
  const _Float16* arow = attnP + ((size_t)b * HW + pbase + ln) * C_DIM;

  // Hoist all 8 B-fragments (shared by every channel tile of this wave).
  AF bfr[8];
  for (int ks = 0; ks < 8; ++ks) {
    const _Float16* ar = arow + ks * 32 + hl * 8;
    bfr[ks].f4[0] = *(const f32x4*)(ar);
    bfr[ks].f4[1] = *(const f32x4*)(ar + 16);
  }

  for (int i = 0; i < 4; ++i) {                    // 4 channel-tiles per wave
    int obase = (wave * 4 + i) * 16;
    v8f acc = {};
    for (int ks = 0; ks < 8; ++ks) {
      AF a;
      const _Float16* wr = Wpf16 + (size_t)(obase + ln) * C_DIM + ks * 32 + hl * 8;
      a.f4[0] = *(const f32x4*)(wr);
      a.f4[1] = *(const f32x4*)(wr + 16);
      acc = __builtin_amdgcn_wmma_f32_16x16x32_f16(false, a.v, false, bfr[ks].v,
                                                   (short)0, acc, false, false);
    }
    for (int r = 0; r < 8; ++r) {                  // + bias + residual -> LDS
      int o = obase + r + 8 * hl;
      int pix = pbase + ln;
      ybuf[ln * 264 + o] = acc[r] + bp[o] + x[((size_t)b * C_DIM + o) * HW + pix];
    }
  }
  __syncthreads();

  int p = threadIdx.x >> 3;                        // pixel 0..15
  int ch0 = (threadIdx.x & 7) * 32;                // 32-channel slice
  float s = 0.f, sq = 0.f;
  for (int c = 0; c < 32; ++c) {
    float v = ybuf[p * 264 + ch0 + c];
    s += v; sq += v * v;
  }
  psum[threadIdx.x] = s; psq[threadIdx.x] = sq;
  __syncthreads();
  if (threadIdx.x < 16) {
    float S = 0.f, SQ = 0.f;
    for (int k = 0; k < 8; ++k) { S += psum[threadIdx.x * 8 + k]; SQ += psq[threadIdx.x * 8 + k]; }
    float mu = S * (1.0f / 256.0f);
    float var = SQ * (1.0f / 256.0f) - mu * mu;
    mean_s[threadIdx.x] = mu;
    inv_s[threadIdx.x] = rsqrtf(var + 1e-5f);
  }
  __syncthreads();
  float mu = mean_s[p], iv = inv_s[p];
  for (int c = 0; c < 32; ++c) {
    int ch = ch0 + c;
    float v = (ybuf[p * 264 + ch] - mu) * iv * gamma[ch] + beta[ch];
    out[((size_t)b * C_DIM + ch) * HW + pbase + p] = v;
  }
}

// ---------------------------------------------------------------------------
extern "C" void kernel_launch(void* const* d_in, const int* in_sizes, int n_in,
                              void* d_out, int out_size, void* d_ws, size_t ws_size,
                              hipStream_t stream) {
  (void)in_sizes; (void)n_in; (void)out_size; (void)ws_size;
  const float* x     = (const float*)d_in[0];
  const float* Wq    = (const float*)d_in[1];
  const float* bq    = (const float*)d_in[2];
  const float* Wk    = (const float*)d_in[3];
  const float* bk    = (const float*)d_in[4];
  const float* Wv    = (const float*)d_in[5];
  const float* bv    = (const float*)d_in[6];
  const float* Wp    = (const float*)d_in[7];
  const float* bp    = (const float*)d_in[8];
  const float* gamma = (const float*)d_in[9];
  const float* beta  = (const float*)d_in[10];
  float* out = (float*)d_out;

  char* ws = (char*)d_ws;
  _Float16* Wf16  = (_Float16*)ws;                              // 512 KB
  _Float16* xt    = (_Float16*)(ws + 524288);                   // 4 MB
  _Float16* qt    = (_Float16*)(ws + 524288 + 1 * 4194304);     // 4 MB
  _Float16* kt    = (_Float16*)(ws + 524288 + 2 * 4194304);     // 4 MB
  _Float16* vdm   = (_Float16*)(ws + 524288 + 3 * 4194304);     // 4 MB
  _Float16* attnP = (_Float16*)(ws + 524288 + 4 * 4194304);     // 4 MB

  prep_weights<<<1024, 256, 0, stream>>>(Wq, Wk, Wv, Wp, Wf16);
  prep_x<<<8192, 256, 0, stream>>>(x, xt);
  qkv_kernel<<<1536, 128, 0, stream>>>(Wf16, xt, bq, bk, bv, qt, kt, vdm);
  size_t lds = (size_t)4 * 16 * 1032 * sizeof(_Float16);        // 132096 B / WG
  attn_kernel<<<1024, 128, lds, stream>>>(qt, kt, vdm, attnP);
  proj_ln_kernel<<<512, 128, 0, stream>>>(Wf16 + (size_t)3 * 65536, attnP, bp, x,
                                          gamma, beta, out);
}
